// BoxTracker_10685878633067
// MI455X (gfx1250) — compile-verified, matrix operand order checked by reading, not run on hardware
//
#include <hip/hip_runtime.h>
#include <hip/hip_bf16.h>

// ---------------------------------------------------------------------------
// BoxTracker (ConvLSTM) for gfx1250: all convs as implicit GEMMs on
// v_wmma_f32_16x16x32_bf16 (bf16 A/B, f32 accumulate).
//
// Shapes: T=10, N=4, cin=128, cout=128, gtc=48, H=64, W=64, H2=W2=32
//   z_t = relu(bn(conv_x2h(x_t))) + conv_gt2h(gt_t) + b_gt2h + b_h2h  (all t)
//   step: tmp = z_t + conv_h2h(h_{t-1}); LSTM gates -> c_t, h_t
//         fused GEMM on h_t: rows 0..511 -> hh (next step), 512..559 -> result[t]
//
// Each block covers the FULL M extent for its 64 output columns so the im2col
// gather (B tile) is staged exactly once per column tile.  Weight (A) tiles
// are staged with CDNA5 async global->LDS DMA (GLOBAL_LOAD_ASYNC_TO_LDS_B128,
// ASYNCcnt), overlapping the DMA with the B gather.
// ---------------------------------------------------------------------------

typedef __attribute__((ext_vector_type(16))) __bf16        v16bf;
typedef __attribute__((ext_vector_type(8)))  float         v8f;
typedef __attribute__((ext_vector_type(4)))  unsigned int  v4u;

__device__ __forceinline__ unsigned short f32_to_bf16u(float f) {
  unsigned int u = __float_as_uint(f);
  u += 0x7fffu + ((u >> 16) & 1u);        // round-to-nearest-even
  return (unsigned short)(u >> 16);
}

union FragBF { v4u u[2]; v16bf v; };

// A/B 16-bit fragment layout (ISA 7.12.2): lanes 0-15 hold K 0-7 & 16-23,
// lanes 16-31 hold K 8-15 & 24-31, of the row (A: M=lane%16, B^T: N=lane%16).
__device__ __forceinline__ v16bf load_frag(const unsigned short* rowbase, int hi) {
  FragBF f;
  f.u[0] = *(const v4u*)(rowbase + (hi << 3));        // K 0-7  or 8-15
  f.u[1] = *(const v4u*)(rowbase + 16 + (hi << 3));   // K 16-23 or 24-31
  return f.v;
}

// CDNA5 async global->LDS copy, 16B per lane (ISA 15.18.3 op 98).
// LDS dest address = low 32 bits of the generic pointer (flat LDS aperture:
// low 32 bits are the LDS offset).  Tracked on ASYNCcnt.
__device__ __forceinline__ void async_copy_b128(const void* gsrc, void* ldst) {
  const unsigned lds_off = (unsigned)(size_t)ldst;
  asm volatile("global_load_async_to_lds_b128 %0, %1, off"
               :: "v"(lds_off), "v"(gsrc) : "memory");
}
__device__ __forceinline__ void async_wait_all() {
  asm volatile("s_wait_asynccnt 0" ::: "memory");
}

// ---------------------------------------------------------------------------
// Implicit-GEMM conv kernel.  Block = 256 threads (8 wave32).
// Tile: M = 128*MTILES (full M, zero-padded weights), Ncols = 64, K chunk = 32.
// Wave w owns M-tiles {w, w+8, ..., w+8*(MTILES-1)} (16 rows each).
// MODE 0: x2h   (src f32 x, 128ch 64x64, stride 2)  -> z = relu(bn(acc)) + bias2
// MODE 1: gt2h  (src f32 gt, 48ch 32x32, stride 1, K padded to 448) -> z += acc
// MODE 2: fused h2h+h2gt (src bf16 h, 128ch 32x32, stride 1, M padded to 640)
//         rows<512 -> hh = acc ; rows 512..559 -> result = acc + b_h2gt
// ---------------------------------------------------------------------------
template <int MODE, int MTILES>
__global__ __launch_bounds__(256) void gemm_conv_wmma(
    const unsigned short* __restrict__ Wpk,   // packed bf16 weights [128*MTILES][Kdim]
    const float*          __restrict__ srcF,  // f32 input  (MODE 0,1)
    const unsigned short* __restrict__ srcB,  // bf16 input (MODE 2)
    float*                __restrict__ out0,  // z (0,1) / hh (2)
    float*                __restrict__ out1,  // result_t (2)
    const float* __restrict__ zscale,         // MODE0: bn scale, MODE2: b_h2gt
    const float* __restrict__ zbias,          // MODE0: bn bias
    const float* __restrict__ zbias2,         // MODE0: b_h2h + b_gt2h
    int Mrows, int Kdim)
{
  constexpr int LDA = 40;                     // bf16 units; 80B rows, 16B aligned
  __shared__ unsigned short Alds[128 * MTILES * LDA];  // A tile (128*MTILES) x 32
  __shared__ unsigned short Blds[64 * LDA];            // B^T tile 64 x 32

  const int tid     = threadIdx.x;
  const int wave    = tid >> 5;
  const int lane    = tid & 31;
  const int colbase = blockIdx.x << 6;

  v8f acc[MTILES][4];
  const v8f vzero = {0.f, 0.f, 0.f, 0.f, 0.f, 0.f, 0.f, 0.f};
#pragma unroll
  for (int i = 0; i < MTILES; ++i)
#pragma unroll
    for (int nb = 0; nb < 4; ++nb) acc[i][nb] = vzero;

  // staging assignments
  const int arow = tid >> 1;              // A: 2 threads / row-slab, 16 bf16 each
  const int aseg = (tid & 1) << 4;
  const int bcol = tid & 63;              // B: 4 threads / col, 8 k each
  const int bk0  = (tid >> 6) << 3;

  const int col = colbase + bcol;         // output column -> (img, y, x)
  const int img = col >> 10;
  const int pos = col & 1023;
  const int oy  = pos >> 5;
  const int ox  = pos & 31;

  for (int k0 = 0; k0 < Kdim; k0 += 32) {
    // ---- stage A: async DMA straight into LDS (no VGPR round-trip) ----
#pragma unroll
    for (int i = 0; i < MTILES; ++i) {
      const int grow = (i << 7) + arow;   // local == global row (full-M block)
      const unsigned short* g = Wpk + (size_t)grow * (size_t)Kdim + (k0 + aseg);
      unsigned short*       l = Alds + grow * LDA + aseg;
      async_copy_b128(g, l);              // 8 bf16
      async_copy_b128(g + 8, l + 8);      // 8 bf16
    }
    // ---- stage B^T (im2col gather; consecutive threads -> consecutive x),
    //      overlaps with the async weight DMA above ----
#pragma unroll
    for (int j = 0; j < 8; ++j) {
      const int k = k0 + bk0 + j;
      unsigned short bv = 0;
      if (MODE == 0) {                       // x: 128ch, 64x64, stride 2, pad (0,1)
        const int ci = k / 9, rs = k % 9, r = rs / 3, s = rs % 3;
        const int iy = (oy << 1) + r, ix = (ox << 1) + s;
        if (iy < 64 && ix < 64)
          bv = f32_to_bf16u(srcF[(((size_t)img * 128 + ci) * 64 + iy) * 64 + ix]);
      } else if (MODE == 1) {                // gt: 48ch, 32x32, stride 1, pad 1
        if (k < 432) {
          const int ci = k / 9, rs = k % 9, r = rs / 3, s = rs % 3;
          const int iy = oy + r - 1, ix = ox + s - 1;
          if (iy >= 0 && iy < 32 && ix >= 0 && ix < 32)
            bv = f32_to_bf16u(srcF[(((size_t)img * 48 + ci) * 32 + iy) * 32 + ix]);
        }
      } else {                               // h (bf16): 128ch, 32x32, stride 1
        const int ci = k / 9, rs = k % 9, r = rs / 3, s = rs % 3;
        const int iy = oy + r - 1, ix = ox + s - 1;
        if (iy >= 0 && iy < 32 && ix >= 0 && ix < 32)
          bv = srcB[(((size_t)img * 128 + ci) * 32 + iy) * 32 + ix];
      }
      Blds[bcol * LDA + bk0 + j] = bv;
    }
    async_wait_all();                // drain this wave's async LDS writes
    __syncthreads();

    // ---- compute: 4 B frags reused by MTILES A frags -> 4*MTILES WMMA ----
    const int lrow = lane & 15;
    const int hi   = lane >> 4;
    v16bf bfr[4];
#pragma unroll
    for (int nb = 0; nb < 4; ++nb)
      bfr[nb] = load_frag(Blds + ((nb << 4) + lrow) * LDA, hi);
#pragma unroll
    for (int i = 0; i < MTILES; ++i) {
      const v16bf af =
          load_frag(Alds + (((wave + (i << 3)) << 4) + lrow) * LDA, hi);
#pragma unroll
      for (int nb = 0; nb < 4; ++nb)
        acc[i][nb] = __builtin_amdgcn_wmma_f32_16x16x32_bf16(
            false, af, false, bfr[nb], (short)0, acc[i][nb], false, false);
    }
    __syncthreads();
  }

  // ---- epilogue: C/D layout: lane%16 = N col, VGPR r -> M = r + 8*(lane/16) ----
  const int nc = lane & 15;
#pragma unroll
  for (int i = 0; i < MTILES; ++i) {
    const int mr0 = ((wave + (i << 3)) << 4) + ((lane >> 4) << 3);
#pragma unroll
    for (int nb = 0; nb < 4; ++nb) {
      const int c2  = colbase + (nb << 4) + nc;
      const int im2 = c2 >> 10;
      const int ps2 = c2 & 1023;
#pragma unroll
      for (int r = 0; r < 8; ++r) {
        const int m   = mr0 + r;
        const float v = acc[i][nb][r];
        if (MODE == 0) {
          const size_t o = ((size_t)im2 * 512 + m) * 1024 + ps2;
          out0[o] = fmaxf(v * zscale[m] + zbias[m], 0.f) + zbias2[m];
        } else if (MODE == 1) {
          const size_t o = ((size_t)im2 * 512 + m) * 1024 + ps2;
          out0[o] += v;
        } else {
          if (m < 512) {
            out0[((size_t)im2 * 512 + m) * 1024 + ps2] = v;
          } else if (m < Mrows) {
            out1[((size_t)im2 * 48 + (m - 512)) * 1024 + ps2] = v + zscale[m - 512];
          }
        }
      }
    }
  }
}

// ---------------------------------------------------------------------------
// Elementwise LSTM gate update. idx over N*128*1024 = 524288.
// tmp = z_t (+ hh if t>0); gates split along channel axis in blocks of 128.
// Writes c, h->memory (f32) and h->bf16 (GEMM input for the fused step GEMM).
// ---------------------------------------------------------------------------
__global__ __launch_bounds__(256) void lstm_step_kernel(
    const float* __restrict__ z, const float* __restrict__ hh,
    float* __restrict__ cbuf, unsigned short* __restrict__ hbf,
    float* __restrict__ mem_out, int use_prev)
{
  const int idx = blockIdx.x * 256 + threadIdx.x;     // grid sized exactly
  const int n   = idx >> 17;                          // / (128*1024)
  const int rem = idx & 131071;
  const size_t base = ((size_t)n << 19) + rem;        // n*512*1024 + co*1024 + pos
  float zi = z[base];
  float zf = z[base + 131072];
  float zo = z[base + 262144];
  float zg = z[base + 393216];
  float cp = 0.f;
  if (use_prev) {
    zi += hh[base];
    zf += hh[base + 131072];
    zo += hh[base + 262144];
    zg += hh[base + 393216];
    cp = cbuf[idx];
  }
  const float ig = 1.f / (1.f + expf(-zi));
  const float fg = 1.f / (1.f + expf(-zf));
  const float og = 1.f / (1.f + expf(-zo));
  const float gg = tanhf(zg);
  const float cn = fg * cp + ig * gg;
  const float h  = og * tanhf(cn);
  cbuf[idx]    = cn;
  hbf[idx]     = f32_to_bf16u(h);
  mem_out[idx] = h;
}

// ---------------------------------------------------------------------------
// Weight packing / BN folding
// ---------------------------------------------------------------------------
__global__ __launch_bounds__(256) void pack_f32_bf16(const float* __restrict__ src,
                                                     unsigned short* __restrict__ dst, int n) {
  const int i = blockIdx.x * 256 + threadIdx.x;
  if (i < n) dst[i] = f32_to_bf16u(src[i]);
}

__global__ __launch_bounds__(256) void pack_gt2h_kernel(const float* __restrict__ src,
                                                        unsigned short* __restrict__ dst) {
  const int i = blockIdx.x * 256 + threadIdx.x;      // [512][448], K padded 432->448
  if (i < 512 * 448) {
    const int row = i / 448, k = i - row * 448;
    dst[i] = (k < 432) ? f32_to_bf16u(src[(size_t)row * 432 + k]) : (unsigned short)0;
  }
}

__global__ __launch_bounds__(256) void pack_cat_kernel(const float* __restrict__ wh2h,
                                                       const float* __restrict__ wh2gt,
                                                       unsigned short* __restrict__ dst) {
  const int i = blockIdx.x * 256 + threadIdx.x;      // [640][1152], M padded 560->640
  if (i < 640 * 1152) {
    const int row = i / 1152, k = i - row * 1152;
    unsigned short v = 0;
    if (row < 512)      v = f32_to_bf16u(wh2h[(size_t)row * 1152 + k]);
    else if (row < 560) v = f32_to_bf16u(wh2gt[(size_t)(row - 512) * 1152 + k]);
    dst[i] = v;
  }
}

__global__ __launch_bounds__(256) void make_bn_kernel(
    const float* __restrict__ g, const float* __restrict__ b,
    const float* __restrict__ m, const float* __restrict__ v,
    const float* __restrict__ bh2h, const float* __restrict__ bgt2h,
    float* __restrict__ zs, float* __restrict__ zb, float* __restrict__ zb2) {
  const int i = blockIdx.x * 256 + threadIdx.x;
  if (i < 512) {
    const float s = g[i] * rsqrtf(v[i] + 1e-5f);
    zs[i]  = s;
    zb[i]  = b[i] - m[i] * s;
    zb2[i] = bh2h[i] + bgt2h[i];
  }
}

// ---------------------------------------------------------------------------
extern "C" void kernel_launch(void* const* d_in, const int* in_sizes, int n_in,
                              void* d_out, int out_size, void* d_ws, size_t ws_size,
                              hipStream_t stream) {
  (void)in_sizes; (void)n_in; (void)out_size; (void)ws_size;
  const float* x       = (const float*)d_in[0];   // (10,4,128,64,64)
  const float* prev_gt = (const float*)d_in[1];   // (10,4,48,32,32)
  const float* W_x2h   = (const float*)d_in[2];   // (512,128,3,3)
  const float* bn_g    = (const float*)d_in[3];
  const float* bn_b    = (const float*)d_in[4];
  const float* bn_m    = (const float*)d_in[5];
  const float* bn_v    = (const float*)d_in[6];
  const float* W_h2h   = (const float*)d_in[7];   // (512,128,3,3)
  const float* b_h2h   = (const float*)d_in[8];
  const float* W_gt2h  = (const float*)d_in[9];   // (512,48,3,3)
  const float* b_gt2h  = (const float*)d_in[10];
  const float* W_h2gt  = (const float*)d_in[11];  // (48,128,3,3)
  const float* b_h2gt  = (const float*)d_in[12];

  // workspace carve (all region sizes are multiples of 256 bytes) ~98.6 MB
  char* ws = (char*)d_ws;
  unsigned short* wx2h = (unsigned short*)ws;  ws += (size_t)512 * 1152 * 2;
  unsigned short* wgt  = (unsigned short*)ws;  ws += (size_t)512 * 448 * 2;
  unsigned short* wcat = (unsigned short*)ws;  ws += (size_t)640 * 1152 * 2;  // M-padded
  float* zscale = (float*)ws;                  ws += 2048;
  float* zbias  = (float*)ws;                  ws += 2048;
  float* zbias2 = (float*)ws;                  ws += 2048;
  float* z      = (float*)ws;                  ws += (size_t)40 * 512 * 1024 * 4;
  float* hh     = (float*)ws;                  ws += (size_t)4 * 512 * 1024 * 4;
  float* cbuf   = (float*)ws;                  ws += (size_t)4 * 128 * 1024 * 4;
  unsigned short* hbf = (unsigned short*)ws;   ws += (size_t)4 * 128 * 1024 * 2;

  // --- weight prep (bf16 pack + BN fold) ---
  pack_f32_bf16<<<(512 * 1152 + 255) / 256, 256, 0, stream>>>(W_x2h, wx2h, 512 * 1152);
  pack_gt2h_kernel<<<(512 * 448 + 255) / 256, 256, 0, stream>>>(W_gt2h, wgt);
  pack_cat_kernel<<<(640 * 1152 + 255) / 256, 256, 0, stream>>>(W_h2h, W_h2gt, wcat);
  make_bn_kernel<<<2, 256, 0, stream>>>(bn_g, bn_b, bn_m, bn_v, b_h2h, b_gt2h,
                                        zscale, zbias, zbias2);

  // --- time-batched precompute: z = relu(bn(conv_x2h)) + conv_gt2h + biases ---
  // Full-M blocks (M=512), Ncols = 40*1024 = 40960 -> 640 column tiles
  gemm_conv_wmma<0, 4><<<dim3(640), 256, 0, stream>>>(
      wx2h, x, nullptr, z, nullptr, zscale, zbias, zbias2, 512, 1152);
  gemm_conv_wmma<1, 4><<<dim3(640), 256, 0, stream>>>(
      wgt, prev_gt, nullptr, z, nullptr, nullptr, nullptr, nullptr, 512, 448);

  // --- recurrence: 2 launches / step ---
  float* mem_out = (float*)d_out;                                // (10,4,128,32,32)
  float* res_out = (float*)d_out + (size_t)10 * 4 * 128 * 1024;  // (10,4,48,32,32)
  for (int t = 0; t < 10; ++t) {
    lstm_step_kernel<<<2048, 256, 0, stream>>>(
        z + (size_t)t * 4 * 512 * 1024, hh, cbuf, hbf,
        mem_out + (size_t)t * 4 * 128 * 1024, t > 0 ? 1 : 0);
    // fused [W_h2h ; W_h2gt] GEMM on h_t: M=560 (5 M-tiles/wave), Ncols=4096
    gemm_conv_wmma<2, 5><<<dim3(64), 256, 0, stream>>>(
        wcat, nullptr, hbf, hh, res_out + (size_t)t * 4 * 48 * 1024,
        b_h2gt, nullptr, nullptr, 560, 1152);
  }
}
